// DOMINANTAugmented_36258113913429
// MI455X (gfx1250) — matrix operand order repaired and checked by reading, not compiled, not run on hardware
//
#include <hip/hip_runtime.h>

#define N_NODES 10000
#define N_EDGES 320000
#define IN_DIM  256
#define HID_DIM 64

typedef __attribute__((ext_vector_type(16))) _Float16 v16h;
typedef __attribute__((ext_vector_type(8)))  float    v8f;

// ---------------- utility kernels ----------------

__global__ void k_zero(float* p, long long count) {
  long long i = (long long)blockIdx.x * blockDim.x + threadIdx.x;
  if (i < count) p[i] = 0.0f;
}

__global__ void k_init_deg(float* out_deg, float* in_deg, float* gdeg) {
  int i = blockIdx.x * blockDim.x + threadIdx.x;
  if (i < N_NODES) { out_deg[i] = 0.0f; in_deg[i] = 0.0f; gdeg[i] = 1.0f; /* self loop */ }
}

__global__ void k_deg(const int* __restrict__ ei, float* out_deg, float* in_deg, float* gdeg) {
  int e = blockIdx.x * blockDim.x + threadIdx.x;
  if (e >= N_EDGES) return;
  int s = ei[e], d = ei[N_EDGES + e];
  atomicAdd(&out_deg[s], 1.0f);
  atomicAdd(&in_deg[d], 1.0f);
  atomicAdd(&gdeg[d], 1.0f);
}

__global__ void k_inv(const float* out_deg, const float* in_deg, const float* gdeg,
                      float* oinv, float* iinv, float* dinv) {
  int i = blockIdx.x * blockDim.x + threadIdx.x;
  if (i >= N_NODES) return;
  oinv[i] = 1.0f / fmaxf(out_deg[i], 1e-10f);
  iinv[i] = 1.0f / fmaxf(in_deg[i], 1e-10f);
  dinv[i] = rsqrtf(gdeg[i]);           // gdeg >= 1 always
}

// Bt[n*K + k] = (f16) W[k*Nn + n]   (transpose + convert weights)
__global__ void k_pack_w(const float* __restrict__ W, _Float16* __restrict__ Bt, int K, int Nn) {
  int t = blockIdx.x * blockDim.x + threadIdx.x;
  if (t >= K * Nn) return;
  int n = t / K, k = t - n * K;
  Bt[t] = (_Float16)W[(size_t)k * Nn + n];
}

// per edge (s,d): out_acc[s] += x[d]; in_acc[d] += x[s]
__global__ void k_interp_scatter(const int* __restrict__ ei, const float* __restrict__ x,
                                 float* __restrict__ out_acc, float* __restrict__ in_acc) {
  long long t = (long long)blockIdx.x * blockDim.x + threadIdx.x;
  if (t >= (long long)N_EDGES * IN_DIM) return;
  int e = (int)(t >> 8);
  int f = (int)(t & (IN_DIM - 1));
  int s = ei[e], d = ei[N_EDGES + e];
  atomicAdd(&out_acc[(size_t)s * IN_DIM + f], x[(size_t)d * IN_DIM + f]);
  atomicAdd(&in_acc [(size_t)d * IN_DIM + f], x[(size_t)s * IN_DIM + f]);
}

// x_aug = 0.8*x + 0.1*(out_acc*oinv + in_acc*iinv), cast to f16
__global__ void k_xaug(const float* __restrict__ x, const float* __restrict__ out_acc,
                       const float* __restrict__ in_acc, const float* __restrict__ oinv,
                       const float* __restrict__ iinv, _Float16* __restrict__ xa) {
  long long t = (long long)blockIdx.x * blockDim.x + threadIdx.x;
  if (t >= (long long)N_NODES * IN_DIM) return;
  int i = (int)(t >> 8);
  float v = 0.8f * x[t] + 0.1f * (out_acc[t] * oinv[i] + in_acc[t] * iinv[i]);
  xa[t] = (_Float16)v;
}

// GCN edge aggregation: agg[d] += xw[s] * dinv[s]*dinv[d]
__global__ void k_agg_edges(const int* __restrict__ ei, const float* __restrict__ xw,
                            float* __restrict__ agg, const float* __restrict__ dinv, int flog2) {
  long long t = (long long)blockIdx.x * blockDim.x + threadIdx.x;
  long long total = (long long)N_EDGES << flog2;
  if (t >= total) return;
  int F = 1 << flog2;
  int e = (int)(t >> flog2);
  int f = (int)(t & (F - 1));
  int s = ei[e], d = ei[N_EDGES + e];
  float nrm = dinv[s] * dinv[d];
  atomicAdd(&agg[(size_t)d * F + f], xw[(size_t)s * F + f] * nrm);
}

// add self-loop term + bias, optional relu, write f32 and/or f16
__global__ void k_finish(const float* __restrict__ agg, const float* __restrict__ xw,
                         const float* __restrict__ dinv, const float* __restrict__ bias,
                         float* out32, _Float16* out16, int flog2, int relu) {
  long long t = (long long)blockIdx.x * blockDim.x + threadIdx.x;
  long long total = (long long)N_NODES << flog2;
  if (t >= total) return;
  int F = 1 << flog2;
  int i = (int)(t >> flog2);
  int f = (int)(t & (F - 1));
  float v = agg[t] + xw[t] * dinv[i] * dinv[i] + bias[f];
  if (relu) v = fmaxf(v, 0.0f);
  if (out32) out32[t] = v;
  if (out16) out16[t] = (_Float16)v;
}

// ---------------- WMMA GEMM: C[M,Nn] = A[M,K] * Bt[Nn,K]^T ----------------
// A, Bt: f16 row-major (K contiguous, K % 32 == 0, 16B-aligned rows).
// One wave computes one 16x16 f32 tile via v_wmma_f32_16x16x32_f16.
template <bool NT>
__global__ void k_gemm_nt(const _Float16* __restrict__ A, const _Float16* __restrict__ B,
                          float* __restrict__ C, int M, int Nn, int K,
                          int tilesN, int tilesTotal) {
  int wv = blockIdx.x * (blockDim.x >> 5) + (threadIdx.x >> 5);
  if (wv >= tilesTotal) return;          // wave-uniform: EXEC stays all-ones for WMMA
  int mT = wv / tilesN;
  int nT = wv - mT * tilesN;
  int lane = threadIdx.x & 31;
  int h    = lane >> 4;                  // half-wave id (ISA lane-group 0-15 vs 16-31)
  int idx  = lane & 15;

  const unsigned int* Ad = (const unsigned int*)A;   // dwords = 2 halfs (K even, K+1 odd)
  const unsigned int* Bd = (const unsigned int*)B;
  int Kd = K >> 1;
  size_t arow = (size_t)(mT * 16 + idx) * Kd;
  size_t brow = (size_t)(nT * 16 + idx) * Kd;

  v8f c = {0.f, 0.f, 0.f, 0.f, 0.f, 0.f, 0.f, 0.f};
  int nkb = K >> 5;
  for (int kb = 0; kb < nkb; ++kb) {
    int kd = kb << 4;                    // dword offset of this 32-wide K block
    union { v16h v; uint4 q[2]; } a, b;
    // A 16x32 layout: VGPR v (0..3) K=8h+2v..+1 ; VGPR v (4..7) K=16+8h+2(v-4)..+1
    a.q[0] = *(const uint4*)(Ad + arow + kd + 4 * h);
    a.q[1] = *(const uint4*)(Ad + arow + kd + 4 * h + 8);
    // B 32x16 layout: VGPR v (0..3) K=8h+2v..+1 ; VGPR v (4..7) K=16+8h+2(v-4)..+1
    b.q[0] = *(const uint4*)(Bd + brow + kd + 4 * h);
    b.q[1] = *(const uint4*)(Bd + brow + kd + 4 * h + 8);
    c = __builtin_amdgcn_wmma_f32_16x16x32_f16(false, a.v, false, b.v,
                                               (short)0, c, false, false);
  }

  // C/D layout: VGPR r -> row (mT*16 + r + 8h), col = nT*16 + idx
  size_t col  = (size_t)nT * 16 + idx;
  int    row0 = mT * 16 + 8 * h;
#pragma unroll
  for (int r = 0; r < 8; ++r) {
    float* p = C + (size_t)(row0 + r) * Nn + col;
    if (NT) __builtin_nontemporal_store(c[r], p);
    else    *p = c[r];
  }
}

// ---------------- host launch ----------------

extern "C" void kernel_launch(void* const* d_in, const int* in_sizes, int n_in,
                              void* d_out, int out_size, void* d_ws, size_t ws_size,
                              hipStream_t stream) {
  (void)in_sizes; (void)n_in; (void)out_size; (void)ws_size;
  const float* x      = (const float*)d_in[0];
  const int*   ei     = (const int*)  d_in[1];
  const float* W_enc1 = (const float*)d_in[2];
  const float* b_enc1 = (const float*)d_in[3];
  const float* W_enc2 = (const float*)d_in[4];
  const float* b_enc2 = (const float*)d_in[5];
  const float* W_att1 = (const float*)d_in[6];
  const float* b_att1 = (const float*)d_in[7];
  const float* W_att2 = (const float*)d_in[8];
  const float* b_att2 = (const float*)d_in[9];
  const float* W_str  = (const float*)d_in[10];
  const float* b_str  = (const float*)d_in[11];

  char* ws = (char*)d_ws;
  size_t off = 0;
  auto alloc = [&](size_t bytes) -> char* {
    char* p = ws + off;
    off += (bytes + 255) & ~(size_t)255;
    return p;
  };
  float* bufA    = (float*)alloc((size_t)N_NODES * IN_DIM * 4); // out_acc, then xw
  float* bufB    = (float*)alloc((size_t)N_NODES * IN_DIM * 4); // in_acc, then agg
  float* out_deg = (float*)alloc(N_NODES * 4);
  float* in_deg  = (float*)alloc(N_NODES * 4);
  float* gdeg    = (float*)alloc(N_NODES * 4);
  float* oinv    = (float*)alloc(N_NODES * 4);
  float* iinv    = (float*)alloc(N_NODES * 4);
  float* dinv    = (float*)alloc(N_NODES * 4);
  _Float16* xaug_h = (_Float16*)alloc((size_t)N_NODES * IN_DIM * 2);
  _Float16* h_h    = (_Float16*)alloc((size_t)N_NODES * HID_DIM * 2);
  _Float16* emb_h  = (_Float16*)alloc((size_t)N_NODES * HID_DIM * 2);
  _Float16* h2_h   = (_Float16*)alloc((size_t)N_NODES * HID_DIM * 2);
  _Float16* hs_h   = (_Float16*)alloc((size_t)N_NODES * HID_DIM * 2);
  _Float16* Wp1 = (_Float16*)alloc((size_t)IN_DIM  * HID_DIM * 2); // [64][256]
  _Float16* Wp2 = (_Float16*)alloc((size_t)HID_DIM * HID_DIM * 2); // [64][64]
  _Float16* Wp3 = (_Float16*)alloc((size_t)HID_DIM * HID_DIM * 2);
  _Float16* Wp4 = (_Float16*)alloc((size_t)HID_DIM * IN_DIM  * 2); // [256][64]
  _Float16* Wp5 = (_Float16*)alloc((size_t)HID_DIM * HID_DIM * 2);

  const int T = 256;
  auto nb = [](long long n, int t) { return (unsigned)((n + t - 1) / t); };

  // degrees + interpolation accumulators
  k_init_deg<<<nb(N_NODES, T), T, 0, stream>>>(out_deg, in_deg, gdeg);
  k_zero<<<nb(2LL * N_NODES * IN_DIM, T), T, 0, stream>>>(bufA, 2LL * N_NODES * IN_DIM);
  k_deg<<<nb(N_EDGES, T), T, 0, stream>>>(ei, out_deg, in_deg, gdeg);
  k_inv<<<nb(N_NODES, T), T, 0, stream>>>(out_deg, in_deg, gdeg, oinv, iinv, dinv);

  // weight pack (transpose + f32->f16)
  k_pack_w<<<nb(IN_DIM * HID_DIM, T), T, 0, stream>>>(W_enc1, Wp1, IN_DIM, HID_DIM);
  k_pack_w<<<nb(HID_DIM * HID_DIM, T), T, 0, stream>>>(W_enc2, Wp2, HID_DIM, HID_DIM);
  k_pack_w<<<nb(HID_DIM * HID_DIM, T), T, 0, stream>>>(W_att1, Wp3, HID_DIM, HID_DIM);
  k_pack_w<<<nb(HID_DIM * IN_DIM, T), T, 0, stream>>>(W_att2, Wp4, HID_DIM, IN_DIM);
  k_pack_w<<<nb(HID_DIM * HID_DIM, T), T, 0, stream>>>(W_str, Wp5, HID_DIM, HID_DIM);

  // feature interpolation (sparse, no dense adj)
  k_interp_scatter<<<nb((long long)N_EDGES * IN_DIM, T), T, 0, stream>>>(ei, x, bufA, bufB);
  k_xaug<<<nb((long long)N_NODES * IN_DIM, T), T, 0, stream>>>(x, bufA, bufB, oinv, iinv, xaug_h);

  auto gemm = [&](const _Float16* A, const _Float16* B, float* C,
                  int M, int Nn, int K, bool nt) {
    int tilesN = Nn / 16;
    int tiles = (M / 16) * tilesN;
    unsigned g = (unsigned)((tiles + 7) / 8);   // 8 waves / block
    if (nt) k_gemm_nt<true ><<<g, 256, 0, stream>>>(A, B, C, M, Nn, K, tilesN, tiles);
    else    k_gemm_nt<false><<<g, 256, 0, stream>>>(A, B, C, M, Nn, K, tilesN, tiles);
  };

  auto gcn = [&](const _Float16* Ain, const _Float16* Wp, const float* bias,
                 int K, int F, int flog2, int relu, float* out32, _Float16* out16) {
    gemm(Ain, Wp, bufA, N_NODES, F, K, false);                            // xw = A @ W
    k_zero<<<nb((long long)N_NODES * F, T), T, 0, stream>>>(bufB, (long long)N_NODES * F);
    k_agg_edges<<<nb((long long)N_EDGES * F, T), T, 0, stream>>>(ei, bufA, bufB, dinv, flog2);
    k_finish<<<nb((long long)N_NODES * F, T), T, 0, stream>>>(bufB, bufA, dinv, bias,
                                                              out32, out16, flog2, relu);
  };

  float* x_rec = (float*)d_out;
  float* s_rec = (float*)d_out + (size_t)N_NODES * IN_DIM;

  gcn(xaug_h, Wp1, b_enc1, IN_DIM, HID_DIM, 6, 1, nullptr, h_h);     // enc1 + relu
  gcn(h_h,    Wp2, b_enc2, HID_DIM, HID_DIM, 6, 0, nullptr, emb_h);  // enc2
  gcn(emb_h,  Wp3, b_att1, HID_DIM, HID_DIM, 6, 1, nullptr, h2_h);   // att1 + relu
  gcn(h2_h,   Wp4, b_att2, HID_DIM, IN_DIM,  8, 0, x_rec, nullptr);  // att2 -> x_rec
  gcn(emb_h,  Wp5, b_str,  HID_DIM, HID_DIM, 6, 0, nullptr, hs_h);   // str  -> hs

  // s_rec = hs @ hs^T  (store-bound: 400 MB -> nontemporal stores)
  gemm(hs_h, hs_h, s_rec, N_NODES, N_NODES, HID_DIM, true);
}
